// SwinTransformerBlock2_2087354106340
// MI455X (gfx1250) — matrix operand order repaired and checked by reading, not compiled, not run on hardware
//
#include <hip/hip_runtime.h>
#include <hip/hip_bf16.h>

typedef __bf16 bf16;
typedef __attribute__((ext_vector_type(16))) __bf16 v16bf;
typedef __attribute__((ext_vector_type(8)))  __bf16 v8bf;
typedef __attribute__((ext_vector_type(8)))  float  v8f;

#define B_   16
#define H_   64
#define W_   64
#define C_   384
#define NH_  12
#define HD_  32
#define SS_  4
#define NW_  64
#define BW_  1024          /* B*NW */
#define NT_  128           /* 2N tokens per window-pair */
#define C3_  1152
#define CM_  1536
#define TOK_ 65536         /* B*H*W */
#define SCALE_ 0.17677669529663687f
#define EPS_ 1e-5f
#define LDS_AS 48          /* padded A-tile stride (bf16 elems) */

// ---------------------------------------------------------------- helpers
__device__ __forceinline__ v16bf make_frag(const bf16* lo_p, const bf16* hi_p) {
    v8bf lo = *(const v8bf*)lo_p;
    v8bf hi = *(const v8bf*)hi_p;
    v16bf r;
#pragma unroll
    for (int i = 0; i < 8; ++i) { r[i] = lo[i]; r[i + 8] = hi[i]; }
    return r;
}

// A-fragment (16x32 bf16) from an LDS tile: lane%16 = row, 16 contiguous K
__device__ __forceinline__ v16bf load_a_lds(const bf16* t, int stride, int lane) {
    int m = lane & 15, half = lane >> 4;
    const bf16* p = t + m * stride;
    return make_frag(p + half * 16, p + half * 16 + 8);
}

// B-fragment (32x16 bf16) from row-major W[N][K]: lane%16 = col, 16 contiguous K
__device__ __forceinline__ v16bf load_b_w(const bf16* Wt, int ldk, int n0, int k0, int lane) {
    int n = lane & 15, half = lane >> 4;
    const bf16* p = Wt + (size_t)(n0 + n) * ldk + k0 + half * 16;
    return make_frag(p, p + 8);
}

// CDNA5 async global->LDS copy (16B per lane), tracked by ASYNCcnt.
__device__ __forceinline__ void async_cp16(unsigned int lds_off, const bf16* g) {
    asm volatile("global_load_async_to_lds_b128 %0, %1, off"
                 :: "v"(lds_off), "v"(g) : "memory");
}
__device__ __forceinline__ void wait_async0() {
    asm volatile("s_wait_asynccnt 0x0" ::: "memory");
}

// ---------------------------------------------------------------- weight convert
__global__ __launch_bounds__(256) void cvt_bf16(const float* __restrict__ s,
                                                bf16* __restrict__ d, int n) {
    int i = blockIdx.x * 256 + threadIdx.x;
    if (i < n) d[i] = (bf16)s[i];
}

// ---------------------------------------------------------------- LayerNorm (+optional swin gather)
template <bool GATHER>
__global__ __launch_bounds__(256) void ln_kernel(const float* __restrict__ xa,
                                                 const float* __restrict__ xb,
                                                 const float* __restrict__ gw,
                                                 const float* __restrict__ gb,
                                                 bf16* __restrict__ out, int ntok) {
    int gid = blockIdx.x * 8 + (threadIdx.x >> 5);
    int lane = threadIdx.x & 31;
    if (gid >= ntok) return;
    size_t drow = (size_t)gid * C_;
    const float* src;
    if (GATHER) {
        int bw = gid >> 7, t = gid & 127;
        int s = t >> 6, tt = t & 63;
        int b = bw >> 6, wi = bw & 63;
        int hs = (((wi >> 3) * 8 + (tt >> 3)) + SS_) & (H_ - 1);
        int wsrc = (((wi & 7) * 8 + (tt & 7)) + SS_) & (W_ - 1);
        src = (s ? xb : xa) + (((size_t)b * H_ + hs) * W_ + wsrc) * C_;
    } else {
        src = xa + drow;
    }
    float vals[12];
    float sum = 0.f;
#pragma unroll
    for (int i = 0; i < 12; ++i) { vals[i] = src[lane + 32 * i]; sum += vals[i]; }
#pragma unroll
    for (int m = 16; m >= 1; m >>= 1) sum += __shfl_xor(sum, m, 32);
    float mu = sum * (1.f / (float)C_);
    float vs = 0.f;
#pragma unroll
    for (int i = 0; i < 12; ++i) { float d = vals[i] - mu; vs += d * d; }
#pragma unroll
    for (int m = 16; m >= 1; m >>= 1) vs += __shfl_xor(vs, m, 32);
    float inv = rsqrtf(vs * (1.f / (float)C_) + EPS_);
#pragma unroll
    for (int i = 0; i < 12; ++i) {
        int c = lane + 32 * i;
        out[drow + c] = (bf16)((vals[i] - mu) * inv * gw[c] + gb[c]);
    }
}

// ---------------------------------------------------------------- generic WMMA GEMM
// C[M,N] = A[M,K](bf16) * W[N,K]^T(bf16) + bias, epilogue by MODE.
// MODE 0: split-write scaled q / k / v per head.  MODE 1: proj + reverse/roll + residual.
// MODE 2: exact GELU -> bf16.                     MODE 3: accumulate into f32 out.
// A tile double-buffered in LDS via async global->LDS copies (one barrier / K-step).
template <int MODE>
__global__ __launch_bounds__(256) void gemm_bf16(
    const bf16* __restrict__ A, const bf16* __restrict__ Wt,
    const float* __restrict__ bias, int M, int Nn, int K,
    bf16* __restrict__ ob0, bf16* __restrict__ ob1, bf16* __restrict__ ob2,
    float* __restrict__ of, const float* __restrict__ rx1, const float* __restrict__ rx2) {
    __shared__ __align__(16) bf16 at[2 * 128 * LDS_AS];
    int tid = threadIdx.x;
    int wave = tid >> 5, lane = tid & 31;
    int wm = wave & 3, wn = wave >> 2;
    int m0 = blockIdx.y * 128, n0 = blockIdx.x * 128;

    v8f acc[2][4];
#pragma unroll
    for (int i = 0; i < 2; ++i)
#pragma unroll
        for (int j = 0; j < 4; ++j) acc[i][j] = (v8f)0.f;

    int arow = tid >> 1, ach = tid & 1;
    const bf16* agt = A + (size_t)(m0 + arow) * K + ach * 16;
    unsigned int lbase =
        (unsigned int)(uintptr_t)&at[arow * LDS_AS + ach * 16];
    const int nk = K >> 5;

    // prologue: async-stage K-tile 0 into buffer 0
    async_cp16(lbase, agt);
    async_cp16(lbase + 16, agt + 8);

    for (int it = 0; it < nk; ++it) {
        wait_async0();          // our copies for tile `it` landed
        __syncthreads();        // everyone's copies landed; prev buffer free
        int cur = it & 1;
        if (it + 1 < nk) {      // overlap next tile's copy with this tile's math
            const bf16* an = agt + (it + 1) * 32;
            unsigned int ln = lbase + ((it + 1) & 1) * (128 * LDS_AS * 2);
            async_cp16(ln, an);
            async_cp16(ln + 16, an + 8);
        }
        const bf16* bufc = at + cur * (128 * LDS_AS);
        int k0 = it * 32;

        v16bf af[2], bfr[4];
#pragma unroll
        for (int i = 0; i < 2; ++i)
            af[i] = load_a_lds(bufc + (wm * 32 + i * 16) * LDS_AS, LDS_AS, lane);
#pragma unroll
        for (int j = 0; j < 4; ++j)
            bfr[j] = load_b_w(Wt, K, n0 + wn * 64 + j * 16, k0, lane);
#pragma unroll
        for (int i = 0; i < 2; ++i)
#pragma unroll
            for (int j = 0; j < 4; ++j)
                acc[i][j] = __builtin_amdgcn_wmma_f32_16x16x32_bf16(
                    false, af[i], false, bfr[j], (short)0, acc[i][j], false, false);
    }

    int half = lane >> 4, nn = lane & 15;
#pragma unroll
    for (int i = 0; i < 2; ++i)
#pragma unroll
        for (int j = 0; j < 4; ++j)
#pragma unroll
            for (int r = 0; r < 8; ++r) {
                int row = m0 + wm * 32 + i * 16 + r + 8 * half;
                int col = n0 + wn * 64 + j * 16 + nn;
                float v = acc[i][j][r] + bias[col];
                if (MODE == 0) {
                    int qi = col / C_, jj = col % C_, hd = jj >> 5, d = jj & 31;
                    int bw = row >> 7, t = row & 127;
                    size_t off = (((size_t)bw * NH_ + hd) * NT_ + t) * HD_ + d;
                    if (qi == 0)      ob0[off] = (bf16)(v * SCALE_);
                    else if (qi == 1) ob1[off] = (bf16)v;
                    else              ob2[off] = (bf16)v;
                } else if (MODE == 1) {
                    int bw = row >> 7, t = row & 127;
                    int s = t >> 6, tt = t & 63;
                    int b = bw >> 6, wi = bw & 63;
                    int hf = (((wi >> 3) * 8 + (tt >> 3)) + SS_) & (H_ - 1);
                    int wf = (((wi & 7) * 8 + (tt & 7)) + SS_) & (W_ - 1);
                    size_t pix = (((size_t)b * H_ + hf) * W_ + wf) * C_ + col;
                    const float* rsd = s ? rx2 : rx1;
                    of[(size_t)s * TOK_ * C_ + pix] = rsd[pix] + v;
                } else if (MODE == 2) {
                    float g = 0.5f * v * (1.f + erff(v * 0.70710678118f));
                    ob0[(size_t)row * CM_ + col] = (bf16)g;
                } else {
                    of[(size_t)row * C_ + col] += v;
                }
            }
}

// ---------------------------------------------------------------- attention per (window, head)
__global__ __launch_bounds__(256) void attn_kernel(
    const bf16* __restrict__ qb, const bf16* __restrict__ kb, const bf16* __restrict__ vb,
    const float* __restrict__ rpb, const float* __restrict__ mask,
    bf16* __restrict__ out) {
    __shared__ __align__(16) bf16 qs[128 * 32];
    __shared__ __align__(16) bf16 ks[128 * 32];
    __shared__ __align__(16) bf16 vt[32 * 136];
    __shared__ __align__(16) bf16 ps[128 * 136];

    int blk = blockIdx.x;               // bw*NH + head
    int bw = blk / NH_, head = blk % NH_;
    int wi = bw & 63;
    int tid = threadIdx.x, wave = tid >> 5, lane = tid & 31;
    int half = lane >> 4, nn = lane & 15;
    size_t base = (size_t)blk * NT_ * HD_;

    {   // stage q, k (async direct-to-LDS) and transposed v into LDS
        int tok = tid >> 1, hh = tid & 1;
        const bf16* qp = qb + base + tok * 32 + hh * 16;
        const bf16* kp = kb + base + tok * 32 + hh * 16;
        const bf16* vp = vb + base + tok * 32 + hh * 16;
        unsigned int qdst = (unsigned int)(uintptr_t)&qs[tok * 32 + hh * 16];
        unsigned int kdst = (unsigned int)(uintptr_t)&ks[tok * 32 + hh * 16];
        async_cp16(qdst, qp);
        async_cp16(qdst + 16, qp + 8);
        async_cp16(kdst, kp);
        async_cp16(kdst + 16, kp + 8);
#pragma unroll
        for (int i = 0; i < 16; ++i) vt[(hh * 16 + i) * 136 + tok] = vp[i];
        wait_async0();
    }
    __syncthreads();

    // ---- S = (q*scale) @ k^T : wave owns q rows [wave*16, wave*16+16)
    v8f s[8];
#pragma unroll
    for (int nt = 0; nt < 8; ++nt) s[nt] = (v8f)0.f;
    v16bf aq = load_a_lds(&qs[wave * 16 * 32], 32, lane);
#pragma unroll
    for (int nt = 0; nt < 8; ++nt) {
        const bf16* p = &ks[(nt * 16 + nn) * 32 + half * 16];
        v16bf bk = make_frag(p, p + 8);
        s[nt] = __builtin_amdgcn_wmma_f32_16x16x32_bf16(
            false, aq, false, bk, (short)0, s[nt], false, false);
    }

    // ---- relative-position bias + shift mask
#pragma unroll
    for (int nt = 0; nt < 8; ++nt) {
        int ki = nt * 16 + nn;
        int kq = ki & 63;
        int r2 = kq >> 3, c2 = kq & 7;
#pragma unroll
        for (int r = 0; r < 8; ++r) {
            int qi = wave * 16 + r + 8 * half;
            int qq = qi & 63;
            int ridx = ((qq >> 3) - r2 + 7) * 15 + ((qq & 7) - c2 + 7);
            s[nt][r] += rpb[ridx * NH_ + head] + mask[((size_t)wi * NT_ + qi) * NT_ + ki];
        }
    }

    // ---- softmax along k (distributed over 16 lanes within each lane-half)
#pragma unroll
    for (int r = 0; r < 8; ++r) {
        float mx = -1e30f;
#pragma unroll
        for (int nt = 0; nt < 8; ++nt) mx = fmaxf(mx, s[nt][r]);
#pragma unroll
        for (int m = 1; m < 16; m <<= 1) mx = fmaxf(mx, __shfl_xor(mx, m, 32));
        float sum = 0.f;
#pragma unroll
        for (int nt = 0; nt < 8; ++nt) { float e = __expf(s[nt][r] - mx); s[nt][r] = e; sum += e; }
#pragma unroll
        for (int m = 1; m < 16; m <<= 1) sum += __shfl_xor(sum, m, 32);
        float inv = 1.f / sum;
        int qi = wave * 16 + r + 8 * half;
#pragma unroll
        for (int nt = 0; nt < 8; ++nt)
            ps[qi * 136 + nt * 16 + nn] = (bf16)(s[nt][r] * inv);
    }
    __syncthreads();

    // ---- O = P @ V  (K = 128 tokens, 4 WMMA k-steps, 2 dim-tiles)
    v8f o[2];
    o[0] = (v8f)0.f; o[1] = (v8f)0.f;
#pragma unroll
    for (int kk = 0; kk < 4; ++kk) {
        v16bf ap = load_a_lds(&ps[wave * 16 * 136 + kk * 32], 136, lane);
#pragma unroll
        for (int dt = 0; dt < 2; ++dt) {
            const bf16* p = &vt[(dt * 16 + nn) * 136 + kk * 32 + half * 16];
            v16bf bv = make_frag(p, p + 8);
            o[dt] = __builtin_amdgcn_wmma_f32_16x16x32_bf16(
                false, ap, false, bv, (short)0, o[dt], false, false);
        }
    }
#pragma unroll
    for (int dt = 0; dt < 2; ++dt)
#pragma unroll
        for (int r = 0; r < 8; ++r) {
            int qi = wave * 16 + r + 8 * half;
            out[((size_t)bw * NT_ + qi) * C_ + head * HD_ + dt * 16 + nn] = (bf16)o[dt][r];
        }
}

// ---------------------------------------------------------------- host driver
extern "C" void kernel_launch(void* const* d_in, const int* in_sizes, int n_in,
                              void* d_out, int out_size, void* d_ws, size_t ws_size,
                              hipStream_t stream) {
    const float* x1      = (const float*)d_in[0];
    const float* x2      = (const float*)d_in[1];
    const float* mask    = (const float*)d_in[2];
    const float* n1w     = (const float*)d_in[3];
    const float* n1b     = (const float*)d_in[4];
    const float* rpb     = (const float*)d_in[5];
    const float* qkv_w   = (const float*)d_in[6];
    const float* qkv_b   = (const float*)d_in[7];
    const float* proj_w  = (const float*)d_in[8];
    const float* proj_b  = (const float*)d_in[9];
    const float* n2w     = (const float*)d_in[10];
    const float* n2b     = (const float*)d_in[11];
    const float* fc1_w   = (const float*)d_in[12];
    const float* fc1_b   = (const float*)d_in[13];
    const float* fc2_w   = (const float*)d_in[14];
    const float* fc2_b   = (const float*)d_in[15];
    float* outp = (float*)d_out;

    char* ws = (char*)d_ws;
    const size_t WQ_N = (size_t)C3_ * C_;              // 442368
    const size_t WP_N = (size_t)C_ * C_;               // 147456
    const size_t W1_N = (size_t)CM_ * C_;              // 589824
    const size_t W2_N = (size_t)C_ * CM_;              // 589824
    const size_t XW_N = (size_t)BW_ * NT_ * C_;        // 50,331,648
    const size_t QK_N = (size_t)BW_ * NH_ * NT_ * HD_; // 50,331,648

    bf16* wq   = (bf16*)ws;                 ws += WQ_N * 2;
    bf16* wp   = (bf16*)ws;                 ws += WP_N * 2;
    bf16* w1   = (bf16*)ws;                 ws += W1_N * 2;
    bf16* w2   = (bf16*)ws;                 ws += W2_N * 2;
    bf16* bufA = (bf16*)ws;                 ws += XW_N * 2;  // xw -> attn_out -> xln
    bf16* bufQ = (bf16*)ws;                 ws += QK_N * 2;
    bf16* bufK = (bf16*)ws;                 ws += QK_N * 2;
    bf16* bufV = (bf16*)ws;                 ws += QK_N * 2;
    bf16* hbuf = bufQ;                      // MLP hidden reuses q+k region

    // 1) weights -> bf16
    cvt_bf16<<<(int)((WQ_N + 255) / 256), 256, 0, stream>>>(qkv_w,  wq, (int)WQ_N);
    cvt_bf16<<<(int)((WP_N + 255) / 256), 256, 0, stream>>>(proj_w, wp, (int)WP_N);
    cvt_bf16<<<(int)((W1_N + 255) / 256), 256, 0, stream>>>(fc1_w,  w1, (int)W1_N);
    cvt_bf16<<<(int)((W2_N + 255) / 256), 256, 0, stream>>>(fc2_w,  w2, (int)W2_N);

    // 2) LN + roll(-4) + window partition -> bf16 tokens [BW*NT, C]
    ln_kernel<true><<<(BW_ * NT_) / 8, 256, 0, stream>>>(x1, x2, n1w, n1b, bufA, BW_ * NT_);

    // 3) QKV GEMM -> q(scaled)/k/v per head
    gemm_bf16<0><<<dim3(C3_ / 128, (BW_ * NT_) / 128), 256, 0, stream>>>(
        bufA, wq, qkv_b, BW_ * NT_, C3_, C_, bufQ, bufK, bufV, nullptr, nullptr, nullptr);

    // 4) attention (bias + mask + softmax + PV), writes bf16 [BW*NT, C] into bufA
    attn_kernel<<<BW_ * NH_, 256, 0, stream>>>(bufQ, bufK, bufV, rpb, mask, bufA);

    // 5) proj GEMM + window reverse + roll(+4) + residual -> d_out (f32, both streams)
    gemm_bf16<1><<<dim3(C_ / 128, (BW_ * NT_) / 128), 256, 0, stream>>>(
        bufA, wp, proj_b, BW_ * NT_, C_, C_, nullptr, nullptr, nullptr, outp, x1, x2);

    // 6) MLP per stream: LN -> fc1+GELU -> fc2 += residual
    for (int s = 0; s < 2; ++s) {
        float* xs = outp + (size_t)s * TOK_ * C_;
        ln_kernel<false><<<TOK_ / 8, 256, 0, stream>>>(xs, xs, n2w, n2b, bufA, TOK_);
        gemm_bf16<2><<<dim3(CM_ / 128, TOK_ / 128), 256, 0, stream>>>(
            bufA, w1, fc1_b, TOK_, CM_, C_, hbuf, nullptr, nullptr, nullptr, nullptr, nullptr);
        gemm_bf16<3><<<dim3(C_ / 128, TOK_ / 128), 256, 0, stream>>>(
            hbuf, w2, fc2_b, TOK_, C_, CM_, nullptr, nullptr, nullptr, xs, nullptr, nullptr);
    }
    (void)in_sizes; (void)n_in; (void)out_size; (void)ws_size;
}
